// GraphFusionLayerAtt_36636071035403
// MI455X (gfx1250) — matrix operand order repaired and checked by reading, not compiled, not run on hardware
//
#include <hip/hip_runtime.h>
#include <hip/hip_bf16.h>

typedef _Float16 v16h __attribute__((ext_vector_type(16)));
typedef _Float16 v8h  __attribute__((ext_vector_type(8)));
typedef _Float16 h4   __attribute__((ext_vector_type(4)));
typedef float    v8f  __attribute__((ext_vector_type(8)));

#define H      256
#define TB     32        // batch rows per workgroup
#define NR     64        // node rows per workgroup (2 * TB)
#define NWAVE  8         // 256 threads = 8 wave32

// ---- dynamic LDS layout (bytes) ----
#define OFF_IN  0        // 64x256 f16 (audio rows 0..31, text rows 32..63); reused for fused tile
#define OFF_XN  32768    // 64x256 f16  (stage1 out / stage2 A)
#define OFF_X1  65536    // 64x512 f16  (stage2 out / stage3 A)
#define OFF_X2  131072   // 64x256 f16  (stage3 out)
#define OFF_LG  163840   // 64 f32 logits
#define LDS_BYTES 164096

#define WMMA_F16(a, b, c) \
  __builtin_amdgcn_wmma_f32_16x16x32_f16(false, (a), false, (b), (short)0, (c), false, false)

// ---------------------------------------------------------------------------
// Weight packing: fp32 W[K][N] -> f16 WMMA B-fragments.
// Fragment (nt, kt): lane l holds 16 halves; lane<16: K = kt*32 + 0..7,16..23,
// lane>=16: K = kt*32 + 8..15,24..31; column n = nt*16 + (l&15).
// Stored contiguously: frag index = nt*Ktiles + kt, 32 lanes * 16 halves.
// ---------------------------------------------------------------------------
__global__ void __launch_bounds__(256) pack_weight(const float* __restrict__ W,
                                                   _Float16* __restrict__ out,
                                                   int K, int N) {
  int idx = blockIdx.x * blockDim.x + threadIdx.x;
  if (idx >= K * N) return;
  int j    = idx & 15;
  int lane = (idx >> 4) & 31;
  int tile = idx >> 9;
  int Ktiles = K >> 5;
  int kt = tile % Ktiles;
  int nt = tile / Ktiles;
  int k = (kt << 5) + ((j >> 3) << 4) + ((lane >> 4) << 3) + (j & 7);
  int n = (nt << 4) + (lane & 15);
  out[idx] = (_Float16)W[k * N + n];
}

// ---------------------------------------------------------------------------
// WMMA helpers (layouts per CDNA5 ISA 7.12.2)
// ---------------------------------------------------------------------------
union AFrag { v16h v; v8h h[2]; };

// 16-bit A 16x32: lane<16 row m, K = base+0..7 (vgpr0-3) and base+16..23 (vgpr4-7)
// lane>=16: K shifted by +8.
__device__ __forceinline__ v16h load_a_lds(const _Float16* A, int lda, int mt, int kt, int lane) {
  const int m  = (mt << 4) + (lane & 15);
  const int kb = (kt << 5) + ((lane >> 4) << 3);
  const _Float16* p = A + m * lda + kb;
  AFrag a;
  a.h[0] = *(const v8h*)(p);        // ds_read_b128
  a.h[1] = *(const v8h*)(p + 16);   // ds_read_b128
  return a.v;
}

__device__ __forceinline__ void store_tile_lds(v8f acc, _Float16* __restrict__ D, int N,
                                               int mt, int nt, const float* __restrict__ bias,
                                               bool relu_out, int lane) {
  const int   n  = (nt << 4) + (lane & 15);
  const float bv = bias[n];
  const int   mb = (mt << 4) + ((lane >> 4) << 3);
  _Float16* drow = D + mb * N + n;
#pragma unroll
  for (int v = 0; v < 8; ++v) {
    float x = acc[v] + bv;
    if (relu_out) x = fmaxf(x, 0.0f);
    drow[v * N] = (_Float16)x;
  }
}

// Generic stage with 2x2 register blocking:
// D(f16 LDS) = act( A(f16 LDS) @ W(packed f16 global) + bias(f32) ).
// Each wave computes a 32x32 output block (4 accumulators, 4 independent
// WMMA chains; each A/B fragment feeds 2 WMMAs). Mtiles must be even.
__device__ __forceinline__ void mm_stage_lds(const _Float16* __restrict__ A, int K,
                                             const _Float16* __restrict__ Wp,
                                             const float* __restrict__ bias,
                                             _Float16* __restrict__ D, int N,
                                             int Mtiles, int lgNt, bool relu_out,
                                             int wave, int lane) {
  const int lgNs  = lgNt - 1;                 // n-supertiles = Nt/2
  const int Nsup  = 1 << lgNs;
  const int total = (Mtiles >> 1) << lgNs;    // m-supertiles * n-supertiles
  const int Kt    = K >> 5;
  const v16h* Wv  = (const v16h*)Wp;
  for (int t = wave; t < total; t += NWAVE) {
    const int mt0 = (t >> lgNs) << 1;
    const int nt0 = (t & (Nsup - 1)) << 1;
    v8f acc00 = {}, acc01 = {}, acc10 = {}, acc11 = {};
    const v16h* B0 = Wv + ((nt0 * Kt) << 5) + lane;
    const v16h* B1 = Wv + (((nt0 + 1) * Kt) << 5) + lane;
    for (int kt = 0; kt < Kt; ++kt) {
      v16h a0 = load_a_lds(A, K, mt0,     kt, lane);
      v16h a1 = load_a_lds(A, K, mt0 + 1, kt, lane);
      v16h b0 = *B0;
      v16h b1 = *B1;
      B0 += 32;
      B1 += 32;
      if (kt + 1 < Kt) {
        __builtin_prefetch((const void*)B0, 0, 3);   // near-scope prefetch
        __builtin_prefetch((const void*)B1, 0, 3);
      }
      acc00 = WMMA_F16(a0, b0, acc00);
      acc01 = WMMA_F16(a0, b1, acc01);
      acc10 = WMMA_F16(a1, b0, acc10);
      acc11 = WMMA_F16(a1, b1, acc11);
    }
    store_tile_lds(acc00, D, N, mt0,     nt0,     bias, relu_out, lane);
    store_tile_lds(acc01, D, N, mt0,     nt0 + 1, bias, relu_out, lane);
    store_tile_lds(acc10, D, N, mt0 + 1, nt0,     bias, relu_out, lane);
    store_tile_lds(acc11, D, N, mt0 + 1, nt0 + 1, bias, relu_out, lane);
  }
}

__device__ __forceinline__ void store_tile_out(v8f acc, float* __restrict__ out,
                                               int mt, int nt,
                                               const float* __restrict__ bias, int lane) {
  const int   n  = (nt << 4) + (lane & 15);
  const float bv = bias[n];
  const int   mb = (mt << 4) + ((lane >> 4) << 3);
  float* orow = out + (size_t)mb * H + n;
#pragma unroll
  for (int v = 0; v < 8; ++v) orow[(size_t)v * H] = acc[v] + bv;
}

// Final stage: out(f32 global) = A(f16 LDS, 32x256) @ Wfc + b_fc (M=32,N=256,K=256)
// 2x2 blocking: 1 m-supertile x 8 n-supertiles = 8 supertiles, one per wave.
__device__ __forceinline__ void mm_stage_out(const _Float16* __restrict__ A,
                                             const _Float16* __restrict__ Wp,
                                             const float* __restrict__ bias,
                                             float* __restrict__ out,
                                             int wave, int lane) {
  const int Kt = 8;
  const v16h* Wv = (const v16h*)Wp;
  for (int t = wave; t < 8; t += NWAVE) {
    const int nt0 = t << 1;
    v8f acc00 = {}, acc01 = {}, acc10 = {}, acc11 = {};
    const v16h* B0 = Wv + ((nt0 * Kt) << 5) + lane;
    const v16h* B1 = Wv + (((nt0 + 1) * Kt) << 5) + lane;
    for (int kt = 0; kt < Kt; ++kt) {
      v16h a0 = load_a_lds(A, H, 0, kt, lane);
      v16h a1 = load_a_lds(A, H, 1, kt, lane);
      v16h b0 = *B0;
      v16h b1 = *B1;
      B0 += 32;
      B1 += 32;
      acc00 = WMMA_F16(a0, b0, acc00);
      acc01 = WMMA_F16(a0, b1, acc01);
      acc10 = WMMA_F16(a1, b0, acc10);
      acc11 = WMMA_F16(a1, b1, acc11);
    }
    store_tile_out(acc00, out, 0, nt0,     bias, lane);
    store_tile_out(acc01, out, 0, nt0 + 1, bias, lane);
    store_tile_out(acc10, out, 1, nt0,     bias, lane);
    store_tile_out(acc11, out, 1, nt0 + 1, bias, lane);
  }
}

// ---------------------------------------------------------------------------
// Main fused kernel: proj -> GAT1(identity path) -> GAT2(identity path)
//   -> attention softmax fuse -> FC. 32 batch rows / workgroup.
// ---------------------------------------------------------------------------
__global__ void __launch_bounds__(256) gfa_main(
    const float* __restrict__ audio, const float* __restrict__ text,
    const _Float16* __restrict__ wpa, const float* __restrict__ b_pa,
    const _Float16* __restrict__ wpt, const float* __restrict__ b_pt,
    const _Float16* __restrict__ wg1, const float* __restrict__ b_g1,
    const _Float16* __restrict__ wg2, const float* __restrict__ b_g2,
    const float* __restrict__ W_af, const float* __restrict__ b_af,
    const _Float16* __restrict__ wfc, const float* __restrict__ b_fc,
    float* __restrict__ out) {
  extern __shared__ char smem[];
  _Float16* inb = (_Float16*)(smem + OFF_IN);
  _Float16* xnb = (_Float16*)(smem + OFF_XN);
  _Float16* x1b = (_Float16*)(smem + OFF_X1);
  _Float16* x2b = (_Float16*)(smem + OFF_X2);
  float*    lg  = (float*)(smem + OFF_LG);

  const int tid  = threadIdx.x;
  const int wave = tid >> 5;
  const int lane = tid & 31;
  const int gb0  = blockIdx.x * TB;

  // ---- stage 0: stage inputs into LDS as f16 (32x256 each, contiguous) ----
  const float4* ap = (const float4*)(audio + (size_t)gb0 * H);
  const float4* tp = (const float4*)(text  + (size_t)gb0 * H);
#pragma unroll
  for (int r = 0; r < 8; ++r) {
    int q = tid + (r << 8);                  // float4 index, 0..2047
    float4 va = ap[q];
    float4 vt = tp[q];
    h4 ha = { (_Float16)va.x, (_Float16)va.y, (_Float16)va.z, (_Float16)va.w };
    h4 ht = { (_Float16)vt.x, (_Float16)vt.y, (_Float16)vt.z, (_Float16)vt.w };
    *(h4*)(inb + (q << 2))        = ha;      // audio rows 0..31
    *(h4*)(inb + 8192 + (q << 2)) = ht;      // text  rows 32..63
  }
  __syncthreads();

  // ---- stage 1: xa = relu(A@W_pa+b), xt = relu(T@W_pt+b) ----
  mm_stage_lds(inb,          H, wpa, b_pa, xnb,          H, 2, 4, true, wave, lane);
  mm_stage_lds(inb + 32 * H, H, wpt, b_pt, xnb + 32 * H, H, 2, 4, true, wave, lane);
  __syncthreads();

  // ---- stage 2: x1 = relu(nodes @ W_g1 + b_g1)   [64 x 512] ----
  mm_stage_lds(xnb, H, wg1, b_g1, x1b, 512, 4, 5, true, wave, lane);
  __syncthreads();

  // ---- stage 3: x2 = x1 @ W_g2 + b_g2            [64 x 256] ----
  mm_stage_lds(x1b, 512, wg2, b_g2, x2b, H, 4, 4, false, wave, lane);
  __syncthreads();

  // ---- stage 4: per-node logits x2 @ W_af + b_af ----
  if (tid < NR) {
    const _Float16* xr = x2b + tid * H;
    float s = b_af[0];
    for (int k = 0; k < H; k += 8) {
      v8h xv = *(const v8h*)(xr + k);
#pragma unroll
      for (int u = 0; u < 8; ++u) s += (float)xv[u] * W_af[k + u];
    }
    lg[tid] = s;
  }
  __syncthreads();

  // ---- stage 5: softmax over {audio,text} + weighted fuse -> fused tile ----
  {
    const int i  = tid >> 3;            // batch row in tile (0..31)
    const int cb = (tid & 7) << 5;      // 32-column block
    const float la = lg[i];
    const float lt = lg[32 + i];
    const float mx = fmaxf(la, lt);
    const float ea = __expf(la - mx);
    const float et = __expf(lt - mx);
    const float wa = ea / (ea + et);
    const float wt = et / (ea + et);
    _Float16* fb = inb;                 // input tile is dead: reuse as fused buffer
#pragma unroll 4
    for (int c = cb; c < cb + 32; ++c) {
      float v = wa * (float)x2b[i * H + c] + wt * (float)x2b[(32 + i) * H + c];
      fb[i * H + c] = (_Float16)v;
    }
  }
  __syncthreads();

  // ---- stage 6: out = fused @ W_fc + b_fc ----
  mm_stage_out(inb, wfc, b_fc, out + (size_t)gb0 * H, wave, lane);
}

// ---------------------------------------------------------------------------
// Fixup: full GAT attention path for nodes {0,1} (batch element 0), fp32.
// Overwrites out[0, :]. One workgroup of 256 threads.
// ---------------------------------------------------------------------------
__global__ void __launch_bounds__(256) gat_fixup(
    const float* __restrict__ audio, const float* __restrict__ text,
    const float* __restrict__ W_pa, const float* __restrict__ b_pa,
    const float* __restrict__ W_pt, const float* __restrict__ b_pt,
    const float* __restrict__ W_g1, const float* __restrict__ att_src1,
    const float* __restrict__ att_dst1, const float* __restrict__ b_g1,
    const float* __restrict__ W_g2, const float* __restrict__ att_src2,
    const float* __restrict__ att_dst2, const float* __restrict__ b_g2,
    const float* __restrict__ W_af, const float* __restrict__ b_af,
    const float* __restrict__ W_fc, const float* __restrict__ b_fc,
    float* __restrict__ out) {
  __shared__ float s_n[2][256];
  __shared__ float s_xs[2][512];
  __shared__ float s_x1[2][512];
  __shared__ float s_x2s[2][256];
  __shared__ float s_x2[2][256];
  __shared__ float s_fused[256];
  __shared__ float s_e1[2][2][2];      // [node][head][src(0)/dst(1)]
  __shared__ float s_alpha1[2][2][2];  // [tgt][head][src]
  __shared__ float s_e2[2][2];         // [node][src/dst]
  __shared__ float s_alpha2[2][2];     // [tgt][src]
  __shared__ float s_l[2];

  const int t = threadIdx.x;

  // projections for batch 0
  {
    float s0 = b_pa[t], s1 = b_pt[t];
    for (int k = 0; k < 256; ++k) {
      s0 += audio[k] * W_pa[k * 256 + t];
      s1 += text[k]  * W_pt[k * 256 + t];
    }
    s_n[0][t] = fmaxf(s0, 0.0f);
    s_n[1][t] = fmaxf(s1, 0.0f);
  }
  __syncthreads();

  // GAT1 projection: xs[node][j], j in [0,512)
  for (int node = 0; node < 2; ++node)
    for (int jj = 0; jj < 2; ++jj) {
      int j = t + (jj << 8);
      float s = 0.0f;
      for (int k = 0; k < 256; ++k) s += s_n[node][k] * W_g1[k * 512 + j];
      s_xs[node][j] = s;
    }
  __syncthreads();

  if (t < 8) {  // attention energies, heads=2
    int node = (t >> 2) & 1, h = (t >> 1) & 1, w = t & 1;
    const float* att = w ? att_dst1 : att_src1;
    float s = 0.0f;
    for (int c = 0; c < 256; ++c) s += s_xs[node][h * 256 + c] * att[h * 256 + c];
    s_e1[node][h][w] = s;
  }
  __syncthreads();

  if (t < 4) {  // alpha[tgt][h][src] = softmax_src( lrelu(e_src[src]+e_dst[tgt]) )
    int tgt = t >> 1, h = t & 1;
    float l0 = s_e1[0][h][0] + s_e1[tgt][h][1];
    float l1 = s_e1[1][h][0] + s_e1[tgt][h][1];
    l0 = (l0 > 0.0f) ? l0 : 0.2f * l0;
    l1 = (l1 > 0.0f) ? l1 : 0.2f * l1;
    float mx = fmaxf(l0, l1);
    float e0 = __expf(l0 - mx), e1 = __expf(l1 - mx);
    s_alpha1[tgt][h][0] = e0 / (e0 + e1);
    s_alpha1[tgt][h][1] = e1 / (e0 + e1);
  }
  __syncthreads();

  for (int node = 0; node < 2; ++node)  // aggregate + bias + relu
    for (int jj = 0; jj < 2; ++jj) {
      int j = t + (jj << 8);
      int h = j >> 8;
      float agg = s_alpha1[node][h][0] * s_xs[0][j] + s_alpha1[node][h][1] * s_xs[1][j];
      s_x1[node][j] = fmaxf(agg + b_g1[j], 0.0f);
    }
  __syncthreads();

  // GAT2 projection
  for (int node = 0; node < 2; ++node) {
    float s = 0.0f;
    for (int k = 0; k < 512; ++k) s += s_x1[node][k] * W_g2[k * 256 + t];
    s_x2s[node][t] = s;
  }
  __syncthreads();

  if (t < 4) {
    int node = t >> 1, w = t & 1;
    const float* att = w ? att_dst2 : att_src2;
    float s = 0.0f;
    for (int c = 0; c < 256; ++c) s += s_x2s[node][c] * att[c];
    s_e2[node][w] = s;
  }
  __syncthreads();

  if (t < 2) {
    int tgt = t;
    float l0 = s_e2[0][0] + s_e2[tgt][1];
    float l1 = s_e2[1][0] + s_e2[tgt][1];
    l0 = (l0 > 0.0f) ? l0 : 0.2f * l0;
    l1 = (l1 > 0.0f) ? l1 : 0.2f * l1;
    float mx = fmaxf(l0, l1);
    float e0 = __expf(l0 - mx), e1 = __expf(l1 - mx);
    s_alpha2[tgt][0] = e0 / (e0 + e1);
    s_alpha2[tgt][1] = e1 / (e0 + e1);
  }
  __syncthreads();

  for (int node = 0; node < 2; ++node)
    s_x2[node][t] = s_alpha2[node][0] * s_x2s[0][t] +
                    s_alpha2[node][1] * s_x2s[1][t] + b_g2[t];
  __syncthreads();

  if (t < 2) {  // fusion logits
    float s = b_af[0];
    for (int k = 0; k < 256; ++k) s += s_x2[t][k] * W_af[k];
    s_l[t] = s;
  }
  __syncthreads();

  {
    float mx = fmaxf(s_l[0], s_l[1]);
    float e0 = __expf(s_l[0] - mx), e1 = __expf(s_l[1] - mx);
    float w0 = e0 / (e0 + e1);
    s_fused[t] = w0 * s_x2[0][t] + (1.0f - w0) * s_x2[1][t];
  }
  __syncthreads();

  {
    float s = b_fc[t];
    for (int k = 0; k < 256; ++k) s += s_fused[k] * W_fc[k * 256 + t];
    out[t] = s;
  }
}

// ---------------------------------------------------------------------------
extern "C" void kernel_launch(void* const* d_in, const int* in_sizes, int n_in,
                              void* d_out, int out_size, void* d_ws, size_t ws_size,
                              hipStream_t stream) {
  const float* audio    = (const float*)d_in[0];
  const float* text     = (const float*)d_in[1];
  const float* W_pa     = (const float*)d_in[2];
  const float* b_pa     = (const float*)d_in[3];
  const float* W_pt     = (const float*)d_in[4];
  const float* b_pt     = (const float*)d_in[5];
  const float* W_g1     = (const float*)d_in[6];
  const float* att_src1 = (const float*)d_in[7];
  const float* att_dst1 = (const float*)d_in[8];
  const float* b_g1     = (const float*)d_in[9];
  const float* W_g2     = (const float*)d_in[10];
  const float* att_src2 = (const float*)d_in[11];
  const float* att_dst2 = (const float*)d_in[12];
  const float* b_g2     = (const float*)d_in[13];
  const float* W_af     = (const float*)d_in[14];
  const float* b_af     = (const float*)d_in[15];
  const float* W_fc     = (const float*)d_in[16];
  const float* b_fc     = (const float*)d_in[17];
  float* out = (float*)d_out;

  // packed f16 weights in workspace (halves): total 458752 halves = 896 KiB
  _Float16* ws  = (_Float16*)d_ws;
  _Float16* wpa = ws;
  _Float16* wpt = ws + 65536;
  _Float16* wg1 = ws + 131072;
  _Float16* wg2 = ws + 262144;
  _Float16* wfc = ws + 393216;

  pack_weight<<<256, 256, 0, stream>>>(W_pa, wpa, 256, 256);
  pack_weight<<<256, 256, 0, stream>>>(W_pt, wpt, 256, 256);
  pack_weight<<<512, 256, 0, stream>>>(W_g1, wg1, 256, 512);
  pack_weight<<<512, 256, 0, stream>>>(W_g2, wg2, 512, 256);
  pack_weight<<<256, 256, 0, stream>>>(W_fc, wfc, 256, 256);

  const int Bn   = in_sizes[0] / H;   // 65536
  const int grid = Bn / TB;           // 2048

  gfa_main<<<grid, 256, LDS_BYTES, stream>>>(audio, text, wpa, b_pa, wpt, b_pt,
                                             wg1, b_g1, wg2, b_g2,
                                             W_af, b_af, wfc, b_fc, out);

  gat_fixup<<<1, 256, 0, stream>>>(audio, text, W_pa, b_pa, W_pt, b_pt,
                                   W_g1, att_src1, att_dst1, b_g1,
                                   W_g2, att_src2, att_dst2, b_g2,
                                   W_af, b_af, W_fc, b_fc, out);
}
